// ResNet_88665304858677
// MI455X (gfx1250) — matrix operand order repaired and checked by reading, not compiled, not run on hardware
//
#include <hip/hip_runtime.h>

// ---------------------------------------------------------------------------
// MI455X (gfx1250) spiking-ResNet forward.
// Convolutions are implicit-GEMM on V_WMMA_F32_16X16X32_F16 (wave32).
// K-tile = 64 (4 WMMAs per stage/barrier pair). Global->register gathers are
// software-pipelined one K-step ahead; out-of-bounds gathers are redirected
// to a zero guard page so no predicates live across pipeline stages.
// Weights are f16 with K padded to 64 -> unconditional aligned b32 copies.
// ---------------------------------------------------------------------------

typedef _Float16 v16h __attribute__((ext_vector_type(16)));
typedef _Float16 h8   __attribute__((ext_vector_type(8)));
typedef _Float16 h2   __attribute__((ext_vector_type(2)));
typedef float    v8f  __attribute__((ext_vector_type(8)));

#define GS_FOR(i, n) for (int i = blockIdx.x * blockDim.x + threadIdx.x; i < (n); i += gridDim.x * blockDim.x)

// ----------------------------- elementwise ---------------------------------
__global__ void k_zero(float* p, int n) { GS_FOR(i, n) p[i] = 0.f; }

// f32 OIHW weights -> f16 with K-dim padded (zero-filled) to KKp
__global__ void k_w16pad(const float* __restrict__ src, _Float16* __restrict__ dst,
                         int KK, int KKp) {
  int n = blockIdx.y;
  for (int k = blockIdx.x * 256 + threadIdx.x; k < KKp; k += gridDim.x * 256)
    dst[(size_t)n * KKp + k] = (k < KK) ? (_Float16)src[(size_t)n * KK + k] : (_Float16)0.f;
}

// scale = gamma/sqrt(var+eps) ; bias = beta - mean*scale + conv_bias*scale
__global__ void k_fold_bn(const float* __restrict__ g, const float* __restrict__ bt,
                          const float* __restrict__ mn, const float* __restrict__ vr,
                          const float* __restrict__ cb, float* __restrict__ sc,
                          float* __restrict__ bi, int C) {
  int i = blockIdx.x * blockDim.x + threadIdx.x;
  if (i < C) {
    float inv = g[i] * rsqrtf(vr[i] + 1e-5f);
    sc[i] = inv;
    bi[i] = bt[i] - mn[i] * inv + cb[i] * inv;
  }
}

__global__ void k_relu(const float* __restrict__ s, float* __restrict__ d, int n) {
  GS_FOR(i, n) d[i] = fmaxf(s[i], 0.f);
}

__global__ void k_add_relu(const float* __restrict__ a, const float* __restrict__ b,
                           float* __restrict__ d, int n) {
  GS_FOR(i, n) d[i] = fmaxf(a[i] + b[i], 0.f);
}

// IF neuron: m += I ; s = (m >= 1) ; m -= s   (soft reset, VTH = 1.0)
__global__ void k_if(float* __restrict__ mem, const float* __restrict__ cur,
                     float* __restrict__ spk, int n) {
  GS_FOR(i, n) {
    float m = mem[i] + cur[i];
    float s = (m >= 1.f) ? 1.f : 0.f;
    mem[i] = m - s;
    spk[i] = s;
  }
}

__global__ void k_if2(float* __restrict__ mem, const float* __restrict__ cur,
                      const float* __restrict__ res, float* __restrict__ spk, int n) {
  GS_FOR(i, n) {
    float m = mem[i] + cur[i] + res[i];
    float s = (m >= 1.f) ? 1.f : 0.f;
    mem[i] = m - s;
    spk[i] = s;
  }
}

__global__ void k_scale(const float* __restrict__ s, float* __restrict__ d, float a, int n) {
  GS_FOR(i, n) d[i] = s[i] * a;
}

// ----------------------------- FC / logits ---------------------------------
__global__ __launch_bounds__(256) void k_fc(const float* __restrict__ in,
                                            const float* __restrict__ w,
                                            const float* __restrict__ bias,
                                            float* __restrict__ out,
                                            int K, int nclass, float alpha, int accumulate) {
  int b = blockIdx.x, c = blockIdx.y;
  const float* xr = in + (size_t)b * K;
  const float* wr = w + (size_t)c * K;
  float s = 0.f;
  for (int k = threadIdx.x; k < K; k += 256) s += xr[k] * wr[k];
  __shared__ float red[256];
  red[threadIdx.x] = s;
  __syncthreads();
  for (int off = 128; off > 0; off >>= 1) {
    if (threadIdx.x < off) red[threadIdx.x] += red[threadIdx.x + off];
    __syncthreads();
  }
  if (threadIdx.x == 0) {
    float v = alpha * red[0] + bias[c];
    if (accumulate) out[b * nclass + c] += v;
    else            out[b * nclass + c] = v;
  }
}

// --------------------- implicit-GEMM conv + folded BN ----------------------
// A: M x K  (M = B*Ho*Wo, K = Cin*KD*KD)  im2col gather, f32 -> f16
// B: N x K  (N = Cout)                     f16 weights, K padded to KKp
// D: NCHW f32, epilogue y = acc*scale[n] + bias[n]
// Block tile 64(M) x 64(N) x 64(K): 8 waves (4 in M x 2 in N),
// 2 K-sub-slices x 2 N-tiles = 4 v_wmma per stage/barrier pair.
// Out-of-bounds gathers load from a zero guard page (address select only);
// gathers for step i+1 overlap the WMMAs of step i.
template <int KD>
__global__ __launch_bounds__(256) void conv_bn_wmma(
    const float* __restrict__ src, const _Float16* __restrict__ wgt,
    const float* __restrict__ scale, const float* __restrict__ bias,
    const float* __restrict__ zpage, float* __restrict__ dst,
    int Bn, int Cin, int hwIn, int wShift, int Cout,
    int woShift, int stride, int pad) {
  const int hwShift = 2 * woShift;
  const int Wo = 1 << woShift;
  const int HW = 1 << hwShift;
  const int Wi = 1 << wShift;
  const int Hi = hwIn >> wShift;
  const int M  = Bn << hwShift;
  const int KK = Cin * KD * KD;
  const int KKp = (KK + 63) & ~63;  // weight K stride (zero padded)

  const int tid  = threadIdx.x;
  const int lane = tid & 31;
  const int wave = tid >> 5;
  const int wm   = wave & 3;   // M sub-tile (16 rows)
  const int wn   = wave >> 2;  // N sub-tile (32 cols -> 2 WMMA tiles)
  const int m0   = blockIdx.x * 64;
  const int n0   = blockIdx.y * 64;

  __shared__ __align__(16) _Float16 As[64 * 72];  // 144B rows (16B aligned)
  __shared__ __align__(16) _Float16 Bs[64 * 72];

  v8f acc[2];
  acc[0] = (v8f){0.f, 0.f, 0.f, 0.f, 0.f, 0.f, 0.f, 0.f};
  acc[1] = acc[0];

  const int l16  = lane & 15;
  const int half = lane >> 4;
  const int rb   = tid >> 5;      // this thread's first staging row
  const int colA = 2 * lane;      // this thread's (even) k-column pair

  // ---- per-thread M-row decode, hoisted out of the K loop ----
  bool mv[8];
  int  base[8], ihb[8], iwb[8];
#pragma unroll
  for (int j = 0; j < 8; j++) {
    int ml = rb + 8 * j;
    int m  = m0 + ml;
    mv[j]  = m < M;
    int b  = m >> hwShift;
    int r  = m & (HW - 1);
    int oh = r >> woShift;
    int ow = r & (Wo - 1);
    ihb[j] = oh * stride - pad;
    iwb[j] = ow * stride - pad;
    base[j] = b * Cin * hwIn;
  }

  // pipeline registers (tile for the *next* stage)
  float va0[8], va1[8];
  unsigned wb[8];

  auto gather = [&](int k0) {
    int kA0 = k0 + colA, kA1 = kA0 + 1;
    int cin0, kh0, kw0, cin1, kh1, kw1;
    if (KD == 3) {
      cin0 = kA0 / 9; { int r = kA0 - cin0 * 9; kh0 = r / 3; kw0 = r - kh0 * 3; }
      cin1 = kA1 / 9; { int r = kA1 - cin1 * 9; kh1 = r / 3; kw1 = r - kh1 * 3; }
    } else {
      cin0 = kA0; kh0 = 0; kw0 = 0;
      cin1 = kA1; kh1 = 0; kw1 = 0;
    }
    const int coff0 = cin0 * hwIn, coff1 = cin1 * hwIn;
    const bool kv0 = kA0 < KK, kv1 = kA1 < KK;
#pragma unroll
    for (int j = 0; j < 8; j++) {
      int ih = ihb[j] + kh0, iw = iwb[j] + kw0;
      bool o = mv[j] && kv0 && (unsigned)ih < (unsigned)Hi && (unsigned)iw < (unsigned)Wi;
      const float* pj = o ? (src + (base[j] + coff0 + (ih << wShift) + iw)) : zpage;
      va0[j] = *pj;  // value already 0 for OOB: no predicate survives the load
      ih = ihb[j] + kh1; iw = iwb[j] + kw1;
      o = mv[j] && kv1 && (unsigned)ih < (unsigned)Hi && (unsigned)iw < (unsigned)Wi;
      pj = o ? (src + (base[j] + coff1 + (ih << wShift) + iw)) : zpage;
      va1[j] = *pj;
    }
#pragma unroll
    for (int j = 0; j < 8; j++)
      wb[j] = *(const unsigned*)&wgt[(size_t)(n0 + rb + 8 * j) * KKp + k0 + colA];
  };

  gather(0);  // prologue

  for (int k0 = 0; k0 < KK; k0 += 64) {
    // ---- drain pipeline regs into LDS (packed h2 / b32 stores) ----
#pragma unroll
    for (int j = 0; j < 8; j++) {
      h2 pk;
      pk[0] = (_Float16)va0[j];
      pk[1] = (_Float16)va1[j];
      *(h2*)&As[(rb + 8 * j) * 72 + colA] = pk;
      *(unsigned*)&Bs[(rb + 8 * j) * 72 + colA] = wb[j];
    }
    __syncthreads();

    // ---- issue next tile's gathers; they overlap the WMMAs below ----
    if (k0 + 64 < KK) {
      gather(k0 + 64);
      if (k0 + 128 < KK)  // prefetch one more weight slice (global_prefetch_b8)
        __builtin_prefetch(&wgt[(size_t)(n0 + (tid & 63)) * KKp + k0 + 128], 0, 1);
    }

    // ---- 2 K-sub-slices x 2 N-tiles = 4 WMMAs ----
#pragma unroll
    for (int s = 0; s < 2; s++) {
      const int cb = s * 32;
      const int arow = (wm * 16 + l16) * 72 + cb;
      h8 alo = *(const h8*)&As[arow + half * 8];
      h8 ahi = *(const h8*)&As[arow + half * 8 + 16];
      v16h a;
#pragma unroll
      for (int i = 0; i < 8; i++) { a[i] = alo[i]; a[i + 8] = ahi[i]; }

      const int kb = half * 8;
      const int bc0 = (wn * 32 + l16) * 72 + cb;
      h8 b0lo = *(const h8*)&Bs[bc0 + kb];
      h8 b0hi = *(const h8*)&Bs[bc0 + kb + 16];
      v16h b0;
#pragma unroll
      for (int i = 0; i < 8; i++) { b0[i] = b0lo[i]; b0[i + 8] = b0hi[i]; }
      const int bc1 = (wn * 32 + 16 + l16) * 72 + cb;
      h8 b1lo = *(const h8*)&Bs[bc1 + kb];
      h8 b1hi = *(const h8*)&Bs[bc1 + kb + 16];
      v16h b1;
#pragma unroll
      for (int i = 0; i < 8; i++) { b1[i] = b1lo[i]; b1[i + 8] = b1hi[i]; }

      acc[0] = __builtin_amdgcn_wmma_f32_16x16x32_f16(false, a, false, b0, (short)0, acc[0], false, false);
      acc[1] = __builtin_amdgcn_wmma_f32_16x16x32_f16(false, a, false, b1, (short)0, acc[1], false, false);
    }
    __syncthreads();
  }

  // ---- epilogue: fold BN, scatter to NCHW (shift-based decode) ----
#pragma unroll
  for (int j = 0; j < 2; j++) {
    int Ncol = n0 + wn * 32 + j * 16 + l16;
    if (Ncol < Cout) {
      float sc = scale[Ncol];
      float bi = bias[Ncol];
#pragma unroll
      for (int r = 0; r < 8; r++) {
        int Mrow = m0 + wm * 16 + half * 8 + r;
        if (Mrow < M) {
          int b  = Mrow >> hwShift;
          int rr = Mrow & (HW - 1);
          dst[(((size_t)(b * Cout + Ncol)) << hwShift) + rr] = acc[j][r] * sc + bi;
        }
      }
    }
  }
}

// ---------------------------------------------------------------------------
// Host side
// ---------------------------------------------------------------------------
namespace {
struct CL {
  const float *w, *cb, *g, *bt, *mn, *vr;
  int cin, cout, k, stride, pad, kkp;
  size_t w16, sc, bi;  // workspace offsets
};
static inline int ilog2(int v) { int s = 0; while ((1 << s) < v) s++; return s; }
}  // namespace

extern "C" void kernel_launch(void* const* d_in, const int* in_sizes, int n_in,
                              void* d_out, int out_size, void* d_ws, size_t ws_size,
                              hipStream_t stream) {
  (void)in_sizes; (void)n_in; (void)out_size; (void)ws_size;
  const int B = 64, STEPS = 5;
  const int cinA[8]   = {64, 64, 64, 128, 128, 256, 256, 512};
  const int coutA[8]  = {64, 64, 128, 128, 256, 256, 512, 512};
  const int strideA[8]= {1, 1, 1, 1, 2, 1, 2, 1};
  const int downA[8]  = {0, 0, 1, 0, 1, 0, 1, 0};
  const int ospA[8]   = {32, 32, 32, 32, 16, 16, 8, 8};

  char* wsb = (char*)d_ws;
  size_t cur = 0;
  auto alloc = [&](size_t bytes) -> size_t {
    size_t o = cur;
    cur += (bytes + 255) & ~(size_t)255;
    return o;
  };

  // ---- parse inputs (dict order of setup_inputs) ----
  int p = 0;
  const float* x = (const float*)d_in[p++];
  CL conv[20];
  int nc = 0;
  auto read_conv = [&](int cin, int cout, int k, int s, int pad) -> int {
    CL c;
    c.w  = (const float*)d_in[p++];
    c.cb = (const float*)d_in[p++];
    c.g  = (const float*)d_in[p++];
    c.bt = (const float*)d_in[p++];
    c.mn = (const float*)d_in[p++];
    c.vr = (const float*)d_in[p++];
    c.cin = cin; c.cout = cout; c.k = k; c.stride = s; c.pad = pad;
    c.kkp = (cin * k * k + 63) & ~63;
    c.w16 = alloc((size_t)cout * c.kkp * sizeof(_Float16));
    c.sc  = alloc((size_t)cout * sizeof(float));
    c.bi  = alloc((size_t)cout * sizeof(float));
    conv[nc] = c;
    return nc++;
  };
  int c0 = read_conv(3, 64, 3, 1, 1);
  int c1i[8], c2i[8], cdi[8];
  for (int i = 0; i < 8; i++) {
    c1i[i] = read_conv(cinA[i], coutA[i], 3, strideA[i], 1);
    c2i[i] = read_conv(coutA[i], coutA[i], 3, 1, 1);
    cdi[i] = downA[i] ? read_conv(cinA[i], coutA[i], 1, strideA[i], 0) : -1;
  }
  const float* fcw = (const float*)d_in[p++];
  const float* fcb = (const float*)d_in[p++];
  // d_in[p] = steps (device scalar, known constant 5)

  // ---- workspace buffers ----
  size_t zpOff = alloc(256);  // zero guard page for OOB gathers
  const int nI0 = B * 64 * 32 * 32;
  size_t I0off = alloc((size_t)nI0 * 4);
  size_t pool[5];
  for (int j = 0; j < 5; j++) pool[j] = alloc((size_t)B * 128 * 32 * 32 * 4);
  size_t memStart = cur;
  size_t mem0 = alloc((size_t)nI0 * 4);
  size_t mem1[8], mem2[8];
  for (int i = 0; i < 8; i++) {
    size_t n = (size_t)B * coutA[i] * ospA[i] * ospA[i] * 4;
    mem1[i] = alloc(n);
    mem2[i] = alloc(n);
  }
  size_t memBytes = cur - memStart;
  size_t logitsOff = alloc(640 * 4);

  float* Pb[5];
  for (int j = 0; j < 5; j++) Pb[j] = (float*)(wsb + pool[j]);
  float* I0 = (float*)(wsb + I0off);
  float* logits = (float*)(wsb + logitsOff);
  float* zpage = (float*)(wsb + zpOff);

  auto ewg = [](int n) { int g = (n + 255) / 256; return g > 8192 ? 8192 : g; };

  // ---- prep: zero guard page, convert weights to padded f16, fold BN ----
  k_zero<<<1, 64, 0, stream>>>(zpage, 64);
  for (int i = 0; i < nc; i++) {
    CL& c = conv[i];
    int KK = c.cin * c.k * c.k;
    dim3 wg((c.kkp + 255) / 256, c.cout);
    k_w16pad<<<wg, 256, 0, stream>>>(c.w, (_Float16*)(wsb + c.w16), KK, c.kkp);
    k_fold_bn<<<(c.cout + 255) / 256, 256, 0, stream>>>(
        c.g, c.bt, c.mn, c.vr, c.cb, (float*)(wsb + c.sc), (float*)(wsb + c.bi), c.cout);
  }

  auto convL = [&](int ci, const float* src, float* dst, int insp, int osp) {
    CL& c = conv[ci];
    int M = B * osp * osp;
    dim3 g((M + 63) / 64, (c.cout + 63) / 64);
    if (c.k == 3)
      conv_bn_wmma<3><<<g, 256, 0, stream>>>(
          src, (const _Float16*)(wsb + c.w16), (const float*)(wsb + c.sc),
          (const float*)(wsb + c.bi), zpage, dst, B, c.cin, insp * insp,
          ilog2(insp), c.cout, ilog2(osp), c.stride, c.pad);
    else
      conv_bn_wmma<1><<<g, 256, 0, stream>>>(
          src, (const _Float16*)(wsb + c.w16), (const float*)(wsb + c.sc),
          (const float*)(wsb + c.bi), zpage, dst, B, c.cin, insp * insp,
          ilog2(insp), c.cout, ilog2(osp), c.stride, c.pad);
  };

  // =========================== ANN path ===========================
  convL(c0, x, I0, 32, 32);  // I0 = bn(conv0(x)) (reused by SNN path)
  int hp = 0;
  float* h = Pb[0];
  k_relu<<<ewg(nI0), 256, 0, stream>>>(I0, h, nI0);
  int insp = 32;
  for (int i = 0; i < 8; i++) {
    int osp = ospA[i];
    int n1 = B * coutA[i] * osp * osp;
    convL(c1i[i], h, Pb[2], insp, osp);
    k_relu<<<ewg(n1), 256, 0, stream>>>(Pb[2], Pb[2], n1);
    convL(c2i[i], Pb[2], Pb[3], osp, osp);
    float* res = h;
    if (cdi[i] >= 0) { convL(cdi[i], h, Pb[4], insp, osp); res = Pb[4]; }
    hp ^= 1;
    k_add_relu<<<ewg(n1), 256, 0, stream>>>(Pb[3], res, Pb[hp], n1);
    h = Pb[hp];
    insp = osp;
  }
  k_fc<<<dim3(B, 10), 256, 0, stream>>>(h, fcw, fcb, (float*)d_out,
                                        8 * 8 * 512, 10, 1.f / STEPS, 0);

  // =========================== SNN path ===========================
  k_zero<<<8192, 256, 0, stream>>>((float*)(wsb + memStart), (int)(memBytes / 4));
  k_zero<<<1, 256, 0, stream>>>(logits, 640);
  for (int t = 0; t < STEPS; t++) {
    hp = 0;
    float* hh = Pb[0];
    k_if<<<ewg(nI0), 256, 0, stream>>>((float*)(wsb + mem0), I0, hh, nI0);
    insp = 32;
    for (int i = 0; i < 8; i++) {
      int osp = ospA[i];
      int n1 = B * coutA[i] * osp * osp;
      convL(c1i[i], hh, Pb[2], insp, osp);
      k_if<<<ewg(n1), 256, 0, stream>>>((float*)(wsb + mem1[i]), Pb[2], Pb[3], n1);
      float* res = hh;
      if (cdi[i] >= 0) { convL(cdi[i], hh, Pb[4], insp, osp); res = Pb[4]; }
      convL(c2i[i], Pb[3], Pb[2], osp, osp);
      hp ^= 1;
      k_if2<<<ewg(n1), 256, 0, stream>>>((float*)(wsb + mem2[i]), Pb[2], res, Pb[hp], n1);
      hh = Pb[hp];
      insp = osp;
    }
    k_fc<<<dim3(B, 10), 256, 0, stream>>>(hh, fcw, fcb, logits, 8 * 8 * 512, 10, 1.f, 1);
  }
  k_scale<<<1, 256, 0, stream>>>(logits, ((float*)d_out) + 640, 1.f / STEPS, 640);
}
